// LVCBlock_79903571575399
// MI455X (gfx1250) — matrix operand order repaired and checked by reading, not compiled, not run on hardware
//
#include <hip/hip_runtime.h>
#include <hip/hip_bf16.h>

typedef __attribute__((ext_vector_type(16))) _Float16 v16h;
typedef __attribute__((ext_vector_type(8)))  float    v8f;

#define B_      16
#define CIN_    32
#define COND_   80
#define T_      128
#define HOP_    256
#define HID_    64
#define LAYERS_ 4
#define L_      (T_ * HOP_)   // 32768
#define COUT_   64            // 2*CIN
#define MROWS_  6144          // CIN*COUT*3 rows of Wk per layer

__device__ __forceinline__ float leakyf(float x, float s) { return x >= 0.f ? x : s * x; }

// Async global->LDS B32 copy (CDNA5, tracked by ASYNCcnt).  Per ISA 10.2 the
// low 32 bits of a flat shared address are the LDS byte offset.
__device__ __forceinline__ void async_g2l_b32(void* lds_dst, const void* gsrc) {
    unsigned int ldst = (unsigned int)(unsigned long long)lds_dst;
    asm volatile("global_load_async_to_lds_b32 %0, %1, off"
                 :: "v"(ldst), "v"(gsrc) : "memory");
}
__device__ __forceinline__ void wait_asynccnt0() {
    asm volatile("s_wait_asynccnt 0x0" ::: "memory");
}

// ---------------------------------------------------------------------------
// Kernel predictor front: input conv (k=5, pad=2) + LeakyReLU(0.1)
// mel (B,80,T) -> c (B,64,T)
// ---------------------------------------------------------------------------
__global__ void k_inconv(const float* __restrict__ mel, const float* __restrict__ Wi,
                         const float* __restrict__ bi, float* __restrict__ c) {
    int idx = blockIdx.x * 256 + threadIdx.x;     // B*HID*T = 131072
    int t = idx & 127, h = (idx >> 7) & 63, b = idx >> 13;
    float acc = bi[h];
    const float* w = Wi + h * COND_ * 5;
    const float* x = mel + b * COND_ * T_;
    for (int ci = 0; ci < COND_; ++ci) {
#pragma unroll
        for (int k = 0; k < 5; ++k) {
            int tt = t + k - 2;
            float xv = (tt >= 0 && tt < T_) ? x[ci * T_ + tt] : 0.f;
            acc += w[ci * 5 + k] * xv;
        }
    }
    c[idx] = leakyf(acc, 0.1f);
}

// Residual stack conv: k=3 pad=1 + LeakyReLU(0.1), 64->64
__global__ void k_resconv(const float* __restrict__ rin, const float* __restrict__ W,
                          const float* __restrict__ bias, float* __restrict__ rout) {
    int idx = blockIdx.x * 256 + threadIdx.x;     // B*HID*T
    int t = idx & 127, h = (idx >> 7) & 63, b = idx >> 13;
    float acc = bias[h];
    const float* w = W + h * HID_ * 3;
    const float* x = rin + b * HID_ * T_;
    for (int ci = 0; ci < HID_; ++ci) {
#pragma unroll
        for (int k = 0; k < 3; ++k) {
            int tt = t + k - 1;
            float xv = (tt >= 0 && tt < T_) ? x[ci * T_ + tt] : 0.f;
            acc += w[ci * 3 + k] * xv;
        }
    }
    rout[idx] = leakyf(acc, 0.1f);
}

__global__ void k_add(float* __restrict__ c, const float* __restrict__ r) {
    int idx = blockIdx.x * 256 + threadIdx.x;
    c[idx] += r[idx];
}

// ---------------------------------------------------------------------------
// Pack one layer's Wk slice (6144 x 192 f32) into WMMA 16x32 f16 A-fragment
// layout: Apack[tileM(384)][ks(6)][lane(32)][slot(16)]
// A-layout (ISA 7.12.2, 16-bit A 16x32): lane = 16*h + m;
//   slot j<8  -> K = 8h + j ; slot j>=8 -> K = 16 + 8h + (j-8)
// ---------------------------------------------------------------------------
__global__ void k_packA(const float* __restrict__ Wk_l, _Float16* __restrict__ Apack) {
    int idx = blockIdx.x * 256 + threadIdx.x;     // 6144*192 = 1179648
    int j = idx & 15;
    int lane = (idx >> 4) & 31;
    int ks = (idx >> 9) % 6;
    int tileM = idx / 3072;
    int h = lane >> 4, m = lane & 15;
    int kloc = (j < 8) ? (8 * h + j) : (16 + 8 * h + (j - 8));
    int kappa = ks * 32 + kloc;                   // = c*3 + jtap, matches Wk row layout
    int rho = tileM * 16 + m;
    Apack[idx] = (_Float16)Wk_l[rho * 192 + kappa];
}

// ---------------------------------------------------------------------------
// Wk GEMM (per batch): out[rho, t] = sum_{c,j} Wk[rho, c*3+j] * cfin[b,c,t+j-1]
// M=6144, N=128, K=192.  8 waves/WG, one 16-row M-tile per wave x full N.
// Epilogue scatters +bk into stage-D A-fragment layout (kernA).
// ---------------------------------------------------------------------------
__global__ void __launch_bounds__(256, 1)
k_wkgemm(const _Float16* __restrict__ Apack, const float* __restrict__ cfin,
         const float* __restrict__ bk_l, _Float16* __restrict__ kernA) {
    __shared__ float ctile[HID_][T_ + 2];                 // padded im2col source
    __shared__ alignas(32) _Float16 bfrag[8][32][16];     // per-k-step B fragments

    int b = blockIdx.y;
    int tid = threadIdx.x;
    int lane = tid & 31, wave = tid >> 5;
    int h = lane >> 4, n = lane & 15;
    int tileM = blockIdx.x * 8 + wave;

    // Interior of the tile via async global->LDS DMA (ASYNCcnt path).
    for (int i = tid; i < HID_ * T_; i += 256) {
        int c = i >> 7, t = i & 127;
        async_g2l_b32(&ctile[c][t + 1], cfin + (size_t)(b * HID_ + c) * T_ + t);
    }
    if (tid < HID_) { ctile[tid][0] = 0.f; ctile[tid][T_ + 1] = 0.f; }
    wait_asynccnt0();

    v8f acc[8] = {};

    for (int ks = 0; ks < 6; ++ks) {
        __syncthreads();
        // Build B fragments (ISA B 32x16 layout: lane=16h+n, slot j<8 -> K=8h+j, j>=8 -> K=16+8h+(j-8))
        for (int i = tid; i < 8 * 32 * 16; i += 256) {
            int j = i & 15, ln = (i >> 4) & 31, nt = i >> 9;
            int hh = ln >> 4, nn = ln & 15;
            int kappa = ks * 32 + ((j < 8) ? (8 * hh + j) : (16 + 8 * hh + (j - 8)));
            int c = kappa / 3, jt = kappa - 3 * c;
            bfrag[nt][ln][j] = (_Float16)ctile[c][nt * 16 + nn + jt];
        }
        __syncthreads();
        v16h a = *(const v16h*)(Apack + (size_t)((tileM * 6 + ks) * 32 + lane) * 16);
#pragma unroll
        for (int nt = 0; nt < 8; ++nt) {
            v16h bv = *(const v16h*)(&bfrag[nt][lane][0]);
            acc[nt] = __builtin_amdgcn_wmma_f32_16x16x32_f16(
                false, a, false, bv, (short)0, acc[nt], false, false);
        }
    }

    // Epilogue: C/D layout row m = 8h + v, col = nt*16 + n.  Scatter into
    // stage-D A-fragment order so the LVC kernel loads contiguous fragments.
    int m8 = 8 * h;
#pragma unroll
    for (int nt = 0; nt < 8; ++nt) {
        int t = nt * 16 + n;
        size_t dbase = ((size_t)(b * T_ + t)) * 12 * 512;
#pragma unroll
        for (int v = 0; v < 8; ++v) {
            int rho = tileM * 16 + m8 + v;
            float val = acc[nt][v] + bk_l[rho];
            int i = rho / 192;
            int rem = rho - i * 192;
            int o = rem / 3, kk = rem - 3 * o;
            int kappa = i * 3 + kk;                 // LVC reduction index
            int tile_o = o >> 4, m = o & 15;
            int kc = kappa >> 5, kloc = kappa & 31;
            int h2, j2;
            if (kloc < 16) { h2 = kloc >> 3; j2 = kloc & 7; }
            else { int r = kloc - 16; h2 = r >> 3; j2 = 8 + (r & 7); }
            int lane2 = m + 16 * h2;
            kernA[dbase + (size_t)(tile_o * 3 + kc) * 512 + lane2 * 16 + j2] = (_Float16)val;
        }
    }
}

// Per-layer bias conv: (B,64,T)  bias_l[b][o][t]
__global__ void k_bias(const float* __restrict__ Wb_l, const float* __restrict__ bb_l,
                       const float* __restrict__ cfin, float* __restrict__ bias_l) {
    int idx = blockIdx.x * 256 + threadIdx.x;     // B*64*T = 131072
    int t = idx & 127, o = (idx >> 7) & 63, b = idx >> 13;
    float acc = bb_l[o];
    const float* w = Wb_l + o * HID_ * 3;
    const float* x = cfin + b * HID_ * T_;
    for (int c = 0; c < HID_; ++c) {
#pragma unroll
        for (int j = 0; j < 3; ++j) {
            int tt = t + j - 1;
            if (tt >= 0 && tt < T_) acc += w[c * 3 + j] * x[c * T_ + tt];
        }
    }
    bias_l[idx] = acc;
}

// ---------------------------------------------------------------------------
// Fused: leaky(0.2) -> dilated conv 32->32 k=3 -> leaky(0.2).  LDS-tiled.
// ---------------------------------------------------------------------------
__global__ void __launch_bounds__(256, 1)
k_dilconv(const float* __restrict__ x, const float* __restrict__ Wc_l,
          const float* __restrict__ bc_l, float* __restrict__ y, int d) {
    __shared__ float xs[CIN_][312];               // 256 + 2*27 max halo
    __shared__ float wt[CIN_][CIN_][3];
    __shared__ float bs[CIN_];
    int b = blockIdx.y, seg = blockIdx.x, tid = threadIdx.x;
    int base = seg * 256;
    int width = 256 + 2 * d;
    for (int i = tid; i < CIN_ * width; i += 256) {
        int ci = i / width, q = i - ci * width;
        int p = base - d + q;
        float v = (p >= 0 && p < L_) ? x[((size_t)b * CIN_ + ci) * L_ + p] : 0.f;
        xs[ci][q] = leakyf(v, 0.2f);
    }
    for (int i = tid; i < CIN_ * CIN_ * 3; i += 256) ((float*)wt)[i] = Wc_l[i];
    if (tid < CIN_) bs[tid] = bc_l[tid];
    __syncthreads();
    int p = tid;
    for (int co = 0; co < CIN_; ++co) {
        float acc = bs[co];
        for (int ci = 0; ci < CIN_; ++ci) {
#pragma unroll
            for (int k = 0; k < 3; ++k) acc += wt[co][ci][k] * xs[ci][p + k * d];
        }
        y[((size_t)b * CIN_ + co) * L_ + base + p] = leakyf(acc, 0.2f);
    }
}

// ---------------------------------------------------------------------------
// LVC + gating, one (b,t) per workgroup.  Per (b,t): y4 = A(64x96) * B(96x256)
// via WMMA f16 tiles; epilogue does sigmoid*tanh gating into 32 channels.
// ---------------------------------------------------------------------------
__global__ void __launch_bounds__(256, 1)
k_lvc(const float* __restrict__ y3, const _Float16* __restrict__ kernA,
      const float* __restrict__ bias_l, float* __restrict__ xout) {
    __shared__ float ytile[CIN_][HOP_ + 4];               // [t*256-1, t*256+257)
    __shared__ alignas(32) _Float16 bfrag[16][32][16];    // all 16 N-tiles, one kc

    int b = blockIdx.y, t = blockIdx.x, tid = threadIdx.x;
    int lane = tid & 31, wave = tid >> 5;
    int h = lane >> 4, n = lane & 15;
    long base = (long)t * HOP_;

    // Interior samples [base, base+256) are always in range: async DMA to LDS.
    for (int i = tid; i < CIN_ * HOP_; i += 256) {
        int ci = i >> 8, s = i & 255;
        async_g2l_b32(&ytile[ci][s + 1], y3 + (size_t)(b * CIN_ + ci) * L_ + base + s);
    }
    // Halo columns (may be zero-padded at the signal edges).
    if (tid < CIN_) {
        long p0 = base - 1, p1 = base + HOP_;
        ytile[tid][0]        = (p0 >= 0) ? y3[((size_t)b * CIN_ + tid) * L_ + p0] : 0.f;
        ytile[tid][HOP_ + 1] = (p1 < L_) ? y3[((size_t)b * CIN_ + tid) * L_ + p1] : 0.f;
    }
    wait_asynccnt0();

    v8f acc[2][4] = {};                                   // [ntl][mt]
    size_t kbase = ((size_t)(b * T_ + t)) * 12 * 512;

    for (int kc = 0; kc < 3; ++kc) {
        __syncthreads();
        for (int i = tid; i < 16 * 32 * 16; i += 256) {
            int j = i & 15, ln = (i >> 4) & 31, nt = i >> 9;
            int hh = ln >> 4, nn = ln & 15;
            int kappa = kc * 32 + ((j < 8) ? (8 * hh + j) : (16 + 8 * hh + (j - 8)));            // = ci*3 + kk
            int ci = kappa / 3, kk = kappa - 3 * ci;
            bfrag[nt][ln][j] = (_Float16)ytile[ci][nt * 16 + nn + kk];
        }
        __syncthreads();
        v16h a[4];
#pragma unroll
        for (int mt = 0; mt < 4; ++mt)
            a[mt] = *(const v16h*)(kernA + kbase + (size_t)(mt * 3 + kc) * 512 + lane * 16);
#pragma unroll
        for (int ntl = 0; ntl < 2; ++ntl) {
            int nt = wave * 2 + ntl;
            v16h bv = *(const v16h*)(&bfrag[nt][lane][0]);
#pragma unroll
            for (int mt = 0; mt < 4; ++mt)
                acc[ntl][mt] = __builtin_amdgcn_wmma_f32_16x16x32_f16(
                    false, a[mt], false, bv, (short)0, acc[ntl][mt], false, false);
        }
    }

    // Gating epilogue: x = sigmoid(y4[:32]) * tanh(y4[32:])
#pragma unroll
    for (int ntl = 0; ntl < 2; ++ntl) {
        int s = (wave * 2 + ntl) * 16 + n;
#pragma unroll
        for (int mt = 0; mt < 2; ++mt) {
#pragma unroll
            for (int v = 0; v < 8; ++v) {
                int oc = mt * 16 + 8 * h + v;
                float ga = acc[ntl][mt][v]     + bias_l[(b * COUT_ + oc) * T_ + t];
                float gb = acc[ntl][mt + 2][v] + bias_l[(b * COUT_ + oc + 32) * T_ + t];
                float sg = 1.f / (1.f + __expf(-ga));
                xout[((size_t)b * CIN_ + oc) * L_ + base + s] = sg * tanhf(gb);
            }
        }
    }
}

// ---------------------------------------------------------------------------
extern "C" void kernel_launch(void* const* d_in, const int* in_sizes, int n_in,
                              void* d_out, int out_size, void* d_ws, size_t ws_size,
                              hipStream_t stream) {
    (void)in_sizes; (void)n_in; (void)out_size; (void)ws_size;
    const float* y0   = (const float*)d_in[0];
    const float* mel  = (const float*)d_in[1];
    const float* Wi   = (const float*)d_in[2];
    const float* bi   = (const float*)d_in[3];
    const float* Wres = (const float*)d_in[4];
    const float* bres = (const float*)d_in[5];
    const float* Wk   = (const float*)d_in[6];
    const float* bk   = (const float*)d_in[7];
    const float* Wb   = (const float*)d_in[8];
    const float* bb   = (const float*)d_in[9];
    const float* Wc   = (const float*)d_in[10];
    const float* bc   = (const float*)d_in[11];

    char* ws = (char*)d_ws;
    size_t off = 0;
    auto alloc = [&](size_t bytes) -> void* {
        void* p = ws + off;
        off = (off + bytes + 255) & ~(size_t)255;
        return p;
    };
    float*    c      = (float*)alloc((size_t)B_ * HID_ * T_ * 4);          // 512 KB
    float*    ra     = (float*)alloc((size_t)B_ * HID_ * T_ * 4);
    float*    rb     = (float*)alloc((size_t)B_ * HID_ * T_ * 4);
    _Float16* Apack  = (_Float16*)alloc((size_t)MROWS_ * 192 * 2);         // 2.25 MB
    _Float16* kernA  = (_Float16*)alloc((size_t)B_ * T_ * 12 * 512 * 2);   // 24 MB
    float*    bias_l = (float*)alloc((size_t)B_ * COUT_ * T_ * 4);         // 512 KB
    float*    y3     = (float*)alloc((size_t)B_ * CIN_ * L_ * 4);          // 64 MB
    float*    xA     = (float*)alloc((size_t)B_ * CIN_ * L_ * 4);          // 64 MB
    float*    xB     = (float*)alloc((size_t)B_ * CIN_ * L_ * 4);          // 64 MB

    // ---- kernel predictor front ----
    k_inconv<<<512, 256, 0, stream>>>(mel, Wi, bi, c);
    const float* rin = c;
    float* ro = ra;
    for (int j = 0; j < 6; ++j) {
        k_resconv<<<512, 256, 0, stream>>>(rin, Wres + (size_t)j * HID_ * HID_ * 3,
                                           bres + j * HID_, ro);
        rin = ro;
        ro = (ro == ra) ? rb : ra;
    }
    k_add<<<512, 256, 0, stream>>>(c, rin);   // c <- c + r  (cfin)

    // ---- main LVC stack ----
    const float* xin = y0;
    float* xping[2] = {xA, xB};
    int d = 1;
    for (int layer = 0; layer < LAYERS_; ++layer) {
        k_packA<<<4608, 256, 0, stream>>>(Wk + (size_t)layer * MROWS_ * 192, Apack);
        k_wkgemm<<<dim3(48, B_), 256, 0, stream>>>(Apack, c, bk + layer * MROWS_, kernA);
        k_bias<<<512, 256, 0, stream>>>(Wb + (size_t)layer * COUT_ * HID_ * 3,
                                        bb + layer * COUT_, c, bias_l);
        k_dilconv<<<dim3(L_ / 256, B_), 256, 0, stream>>>(
            xin, Wc + layer * CIN_ * CIN_ * 3, bc + layer * CIN_, y3, d);
        float* xout = (layer == LAYERS_ - 1) ? (float*)d_out : xping[layer & 1];
        k_lvc<<<dim3(T_, B_), 256, 0, stream>>>(y3, kernA, bias_l, xout);
        xin = xout;
        d *= 3;
    }
}
